// ISTFTHead_11647951306838
// MI455X (gfx1250) — matrix operand order repaired and checked by reading, not compiled, not run on hardware
//
#include <hip/hip_runtime.h>

typedef _Float16 f16;
typedef __attribute__((ext_vector_type(16))) _Float16 v16h;
typedef __attribute__((ext_vector_type(8)))  float    v8f;
typedef __attribute__((ext_vector_type(4)))  unsigned int v4u;
typedef __attribute__((ext_vector_type(8)))  int      v8i;
typedef __attribute__((ext_vector_type(4)))  int      v4i;

#define NB    513
#define NFFT  1024
#define HOP   256
#define BATCH 16
#define TFRM  2048
#define MROWS (BATCH * TFRM)          // 32768
#define K1    1024                     // gemm1 K (DIM)
#define N1PAD 1088                     // gemm1 N padded (1026 -> 17*64)
#define K2    1056                     // gemm2 K padded (2*513 -> 33*32)
#define N2    1024                     // gemm2 N (n_fft)
#define OUTLEN 524032                  // hop*(T-1)

#define BLK_M 128
#define BLK_N 64
#define BLK_K 32
#define LDSP  40                       // padded LDS row stride in halves (TDM pad 16B/row)

// ---- workspace layout (bytes) ----
#define OFF_XT    ((size_t)0)                        // f16 [32768,1024] = 67,108,864
#define OFF_WCAT  ((size_t)67108864)                 // f16 [1088,1024]  =  2,228,224
#define OFF_MAG   ((size_t)69337088)                 // f32 [32768,513]  = 67,239,936
#define OFF_INST  ((size_t)136577024)                // f32 [32768,513]  = 67,239,936
#define OFF_B2    ((size_t)203816960)                // f16 [1024,1056]  =  2,162,688
#define OFF_A2    OFF_XT                             // f16 [32768,1056] reuses xt+wcat
#define OFF_FRAMES OFF_MAG                           // f32 [32768,1024] reuses mag+inst
#define WS_NEED   ((size_t)205979648)

// ======================= prep: transpose x and convert =======================
__global__ void prep_x(const float* __restrict__ x, f16* __restrict__ xt) {
  __shared__ float tile[32][33];
  int t0 = blockIdx.x * 32, d0 = blockIdx.y * 32, b = blockIdx.z;
#pragma unroll
  for (int r = 0; r < 4; ++r) {
    int d = d0 + threadIdx.y + r * 8;
    tile[threadIdx.y + r * 8][threadIdx.x] =
        x[((size_t)b * 1024 + d) * 2048 + t0 + threadIdx.x];
  }
  __syncthreads();
#pragma unroll
  for (int r = 0; r < 4; ++r) {
    int t = t0 + threadIdx.y + r * 8;
    xt[((size_t)b * 2048 + t) * 1024 + d0 + threadIdx.x] =
        (f16)tile[threadIdx.x][threadIdx.y + r * 8];
  }
}

__global__ void prep_w(const float* __restrict__ Wm, const float* __restrict__ Wp,
                       f16* __restrict__ wcat) {
  int idx = blockIdx.x * 256 + threadIdx.x;     // < 1088*1024
  int n = idx >> 10, k = idx & 1023;
  float v = 0.f;
  if (n < NB)            v = Wm[n * 1024 + k];
  else if (n < 2 * NB)   v = Wp[(n - NB) * 1024 + k];
  wcat[idx] = (f16)v;
}

// ============== windowed inverse-rfft basis:  B2[n][k], k-major K2 ==========
__global__ void build_basis(f16* __restrict__ B2) {
  int idx = blockIdx.x * 256 + threadIdx.x;     // < 1024*1056
  int n = idx / K2, k = idx - n * K2;
  const float w0 = 6.283185307179586f / 1024.f;
  float win = 0.5f * (1.f - cosf((float)n * w0));
  float v = 0.f;
  if (k < NB) {
    float a = (k == 0 || k == 512) ? 1.f : 2.f;
    int r = (k * n) & 1023;                     // exact angle reduction
    v = a * cosf((float)r * w0) * win * (1.f / 1024.f);
  } else if (k < 2 * NB) {
    int kk = k - NB;
    float a = (kk == 0 || kk == 512) ? 1.f : 2.f;
    int r = (kk * n) & 1023;
    v = -a * sinf((float)r * w0) * win * (1.f / 1024.f);
  }
  B2[idx] = (f16)v;
}

// ================== Tensor Data Mover: 2D tile -> padded LDS ================
// Loads a [rows x 32] f16 tile (row stride = rowStrideElems) into LDS laid out
// with a 40-half row pitch:  pad_interval = 16 DWORDs (64B row),
// pad_amount = 4 DWORDs (16B)  ->  exactly the [rows][LDSP] layout.
__device__ __forceinline__ void tdm_load_tile(const f16* g, int rowStrideElems,
                                              int rows, unsigned ldsByteAddr) {
  unsigned long long ga = (unsigned long long)(uintptr_t)g;
  v4u g0 = { 1u,                                   // count=1 (valid), user mode
             ldsByteAddr,                          // D#.lds_addr
             (unsigned)ga,                         // global_addr[31:0]
             ((unsigned)(ga >> 32) & 0x01ffffffu)  // global_addr[56:32]
                 | (2u << 30) };                   // type = 2 (image)
  v8i g1 = { (int)((1u << 16)                      // data_size = 1 -> 2 bytes
                 | (1u << 20)                      // pad_enable
                 | (3u << 22)                      // pad_interval = 16 DWORDs
                 | (3u << 25)),                    // pad_amount   = 4 DWORDs
             (int)(32u << 16),                     // tensor_dim0[15:0] = 32
             (int)(((unsigned)rows) << 16),        // tensor_dim0 hi=0 | tensor_dim1 lo
             (int)(32u << 16),                     // tensor_dim1 hi=0 | tile_dim0 = 32
             rows,                                 // tile_dim1 | tile_dim2 = 0
             rowStrideElems,                       // tensor_dim0_stride[31:0]
             0, 0 };                               // stride hi / dim1_stride = 0
  v4i z4 = { 0, 0, 0, 0 };
  v8i z8 = { 0, 0, 0, 0, 0, 0, 0, 0 };
  __builtin_amdgcn_tensor_load_to_lds(g0, g1, z4, z4, z8, 0);
}

// =========================== WMMA GEMM core ================================
__device__ __forceinline__ v16h fragA(const f16* rowp, int lane) {
  int kb = (lane >> 4) << 3;   // lanes 0-15: K 0..7,16..23 ; 16-31: 8..15,24..31
  union { uint4 u[2]; v16h v; } f;
  f.u[0] = *reinterpret_cast<const uint4*>(rowp + kb);
  f.u[1] = *reinterpret_cast<const uint4*>(rowp + 16 + kb);
  return f.v;
}
__device__ __forceinline__ v16h fragB(const f16* rowp, int lane) {
  int kb = (lane >> 4) << 3;   // lanes 0-15: K 0..7,16..23 ; 16-31: 8..15,24..31
  union { uint4 u[2]; v16h v; } f;
  f.u[0] = *reinterpret_cast<const uint4*>(rowp + kb);
  f.u[1] = *reinterpret_cast<const uint4*>(rowp + 16 + kb);
  return f.v;
}

// EPILOG 0: mag/inst fused epilogue.  EPILOG 1: plain f32 store (frames).
template <int EPILOG>
__global__ __launch_bounds__(256) void gemm_wmma(
    const f16* __restrict__ A, const f16* __restrict__ Bm, int K, int KT,
    float* __restrict__ out0, float* __restrict__ out1,
    const float* __restrict__ bias0, const float* __restrict__ bias1) {
  __shared__ f16 sA[2][BLK_M * LDSP];
  __shared__ f16 sB[2][BLK_N * LDSP];
  int tid = threadIdx.x, lane = tid & 31, w = tid >> 5;
  int wm = w & 3, wn = w >> 2;                  // 4x2 waves -> 128x64 block tile
  size_t m0 = (size_t)blockIdx.x * BLK_M;
  int n0 = blockIdx.y * BLK_N;
  int wu = __builtin_amdgcn_readfirstlane(w);   // scalar -> guarantees s_cbranch
                                                // (TDM ignores EXEC; must branch)
  unsigned ldsA[2] = { (unsigned)(uintptr_t)&sA[0][0], (unsigned)(uintptr_t)&sA[1][0] };
  unsigned ldsB[2] = { (unsigned)(uintptr_t)&sB[0][0], (unsigned)(uintptr_t)&sB[1][0] };

  v8f acc[2][2] = {};

  // prologue: DMA K-step 0 into buffer 0 (wave0 -> A tile, wave1 -> B tile)
  if (wu == 0)      tdm_load_tile(A + m0 * (size_t)K, K, BLK_M, ldsA[0]);
  else if (wu == 1) tdm_load_tile(Bm + (size_t)n0 * K, K, BLK_N, ldsB[0]);
  __builtin_amdgcn_s_wait_tensorcnt(0);
  __syncthreads();

  for (int kt = 0; kt < KT; ++kt) {
    int cur = kt & 1;
    if (kt + 1 < KT) {                          // overlap next DMA with WMMAs
      int k0 = (kt + 1) * BLK_K;
      if (wu == 0)      tdm_load_tile(A + m0 * (size_t)K + k0, K, BLK_M, ldsA[cur ^ 1]);
      else if (wu == 1) tdm_load_tile(Bm + (size_t)n0 * K + k0, K, BLK_N, ldsB[cur ^ 1]);
    }
#pragma unroll
    for (int m16 = 0; m16 < 2; ++m16) {
      int arow = wm * 32 + m16 * 16 + (lane & 15);
      v16h af = fragA(&sA[cur][arow * LDSP], lane);
#pragma unroll
      for (int n16 = 0; n16 < 2; ++n16) {
        int brow = wn * 32 + n16 * 16 + (lane & 15);
        v16h bf = fragB(&sB[cur][brow * LDSP], lane);
        acc[m16][n16] = __builtin_amdgcn_wmma_f32_16x16x32_f16(
            false, af, false, bf, (short)0, acc[m16][n16], false, false);
      }
    }
    __builtin_amdgcn_s_wait_tensorcnt(0);       // issuer: next buffer landed
    __syncthreads();                            // publish to all waves
  }

  // epilogue (C/D layout: lanes 0-15 -> M=j, N=lane; lanes 16-31 -> M=8+j, N=lane-16)
  int cB = lane & 15;
  int rB = (lane >> 4) * 8;
#pragma unroll
  for (int m16 = 0; m16 < 2; ++m16)
#pragma unroll
    for (int n16 = 0; n16 < 2; ++n16)
#pragma unroll
      for (int j = 0; j < 8; ++j) {
        size_t r = m0 + wm * 32 + m16 * 16 + rB + j;
        int cg = n0 + wn * 32 + n16 * 16 + cB;
        float v = acc[m16][n16][j];
        if (EPILOG == 0) {
          if (cg < NB) {
            float lm = v + bias0[cg];
            lm = fminf(fmaxf(lm, -10.f), 10.f);
            out0[r * NB + cg] = expf(lm);
          } else if (cg < 2 * NB) {
            int cc = cg - NB;
            float ph = v + bias1[cc];
            out1[r * NB + cc] = fminf(fmaxf(ph, -3.14159265358979f), 3.14159265358979f);
          }
        } else {
          out0[r * (size_t)N2 + cg] = v;
        }
      }
}

// ================= cumsum over time + polar->cartesian (f16 A2) =============
__global__ void cumsum_phase(const float* __restrict__ mag,
                             const float* __restrict__ inst,
                             f16* __restrict__ A2) {
  int b = blockIdx.x, k = threadIdx.x;          // 544 threads
  size_t base = (size_t)b * TFRM * NB;
  size_t abase = (size_t)b * TFRM * K2;
  if (k < NB) {
    float phase = 0.f;
    for (int t = 0; t < TFRM; ++t) {
      phase += inst[base + (size_t)t * NB + k];
      float m = mag[base + (size_t)t * NB + k];
      float s, c;
      sincosf(phase, &s, &c);
      A2[abase + (size_t)t * K2 + k]      = (f16)(m * c);
      A2[abase + (size_t)t * K2 + NB + k] = (f16)(m * s);
    }
  } else if (k < NB + 30) {                     // zero the K padding cols 1026..1055
    int col = 2 * NB + (k - NB);
    for (int t = 0; t < TFRM; ++t)
      A2[abase + (size_t)t * K2 + col] = (f16)0.f;
  }
}

// ============= overlap-add + envelope divide + clip (deterministic) =========
__global__ void ola_kernel(const float* __restrict__ frames, float* __restrict__ out) {
  int idx = blockIdx.x * 256 + threadIdx.x;
  if (idx >= BATCH * OUTLEN) return;
  int b = idx / OUTLEN;
  int i = idx - b * OUTLEN;
  int p = i + NFFT / 2;
  int t_hi = p >> 8; if (t_hi > TFRM - 1) t_hi = TFRM - 1;
  int t_lo = (p - 768) >> 8; if (t_lo < 0) t_lo = 0;  // ceil((p-1023)/256)
  const float w0 = 6.283185307179586f / 1024.f;
  float sig = 0.f, env = 0.f;
  for (int t = t_lo; t <= t_hi; ++t) {
    int n = p - (t << 8);
    float w = 0.5f * (1.f - cosf((float)n * w0));
    sig += frames[((size_t)(b * TFRM + t)) * NFFT + n];
    env += w * w;
  }
  float a = sig / env;
  out[idx] = fminf(fmaxf(a, -1.f), 1.f);
}

// ================================ launcher ==================================
extern "C" void kernel_launch(void* const* d_in, const int* in_sizes, int n_in,
                              void* d_out, int out_size, void* d_ws, size_t ws_size,
                              hipStream_t stream) {
  (void)in_sizes; (void)n_in; (void)out_size;
  if (ws_size < WS_NEED) return;
  const float* x       = (const float*)d_in[0];
  const float* W_mag   = (const float*)d_in[1];
  const float* b_mag   = (const float*)d_in[2];
  const float* W_phase = (const float*)d_in[3];
  const float* b_phase = (const float*)d_in[4];

  char* ws = (char*)d_ws;
  f16*   xt     = (f16*)(ws + OFF_XT);
  f16*   wcat   = (f16*)(ws + OFF_WCAT);
  float* mag    = (float*)(ws + OFF_MAG);
  float* inst   = (float*)(ws + OFF_INST);
  f16*   A2     = (f16*)(ws + OFF_A2);
  f16*   B2     = (f16*)(ws + OFF_B2);
  float* frames = (float*)(ws + OFF_FRAMES);
  float* out    = (float*)d_out;

  prep_x<<<dim3(TFRM / 32, 1024 / 32, BATCH), dim3(32, 8), 0, stream>>>(x, xt);
  prep_w<<<(N1PAD * 1024) / 256, 256, 0, stream>>>(W_mag, W_phase, wcat);
  build_basis<<<(N2 * K2) / 256, 256, 0, stream>>>(B2);

  gemm_wmma<0><<<dim3(MROWS / BLK_M, N1PAD / BLK_N), 256, 0, stream>>>(
      xt, wcat, K1, K1 / BLK_K, mag, inst, b_mag, b_phase);

  cumsum_phase<<<BATCH, 544, 0, stream>>>(mag, inst, A2);

  gemm_wmma<1><<<dim3(MROWS / BLK_M, N2 / BLK_N), 256, 0, stream>>>(
      A2, B2, K2, K2 / BLK_K, frames, nullptr, nullptr, nullptr);

  ola_kernel<<<(BATCH * OUTLEN + 255) / 256, 256, 0, stream>>>(frames, out);
}
